// GaussianRenderer_5016521802262
// MI455X (gfx1250) — compile-verified, matrix operand order checked by reading, not run on hardware
//
#include <hip/hip_runtime.h>
#include <hip/hip_bf16.h>
#include <cstdint>
#include <cstddef>

typedef __attribute__((ext_vector_type(16))) _Float16     v16h;
typedef __attribute__((ext_vector_type(8)))  float        v8f;
typedef __attribute__((ext_vector_type(4)))  float        v4f;
typedef __attribute__((ext_vector_type(4)))  unsigned int v4u;
typedef __attribute__((ext_vector_type(8)))  unsigned int v8u;
typedef __attribute__((ext_vector_type(2)))  _Float16     half2v;

#define NGAUSS  512
#define NFEAT   32
#define NPOINTS (96 * 96 * 16)
#define NCHUNK  (NGAUSS / 32)
#define ROWSTR  20   // dwords per LDS row (80B: 16B aligned, conflict-free)

static __device__ __forceinline__ unsigned int pkh(float a, float b) {
    half2v h;
    h.x = (_Float16)a;
    h.y = (_Float16)b;
    return __builtin_bit_cast(unsigned int, h);
}

// ---------------------------------------------------------------------------
// Per-Gaussian preprocessing: camera transform, quat->R, cov, inverse cov,
// Am = inv_cov * mean, const = mean . Am.  Packed 16 floats per Gaussian.
// ---------------------------------------------------------------------------
__global__ __launch_bounds__(256) void prep_gauss(
    const float* __restrict__ means, const float* __restrict__ scales,
    const float* __restrict__ rots,  const float* __restrict__ opac,
    const float* __restrict__ cam,   float* __restrict__ gp)
{
    int g = blockIdx.x * 256 + threadIdx.x;
    if (g >= NGAUSS) return;

    float mx = means[g * 3 + 0], my = means[g * 3 + 1], mz = means[g * 3 + 2];
    float h0 = cam[0]  * mx + cam[1]  * my + cam[2]  * mz + cam[3];
    float h1 = cam[4]  * mx + cam[5]  * my + cam[6]  * mz + cam[7];
    float hh = cam[8]  * mx + cam[9]  * my + cam[10] * mz + cam[11];
    float h3 = cam[12] * mx + cam[13] * my + cam[14] * mz + cam[15];
    float ih = 1.0f / h3;
    mx = h0 * ih; my = h1 * ih; mz = hh * ih;

    float qw = rots[g * 4 + 0], qx = rots[g * 4 + 1];
    float qy = rots[g * 4 + 2], qz = rots[g * 4 + 3];
    float qn = 1.0f / sqrtf(qw * qw + qx * qx + qy * qy + qz * qz);
    qw *= qn; qx *= qn; qy *= qn; qz *= qn;

    float r00 = 1.f - 2.f * (qy * qy + qz * qz), r01 = 2.f * (qx * qy - qw * qz), r02 = 2.f * (qx * qz + qw * qy);
    float r10 = 2.f * (qx * qy + qw * qz), r11 = 1.f - 2.f * (qx * qx + qz * qz), r12 = 2.f * (qy * qz - qw * qx);
    float r20 = 2.f * (qx * qz - qw * qy), r21 = 2.f * (qy * qz + qw * qx), r22 = 1.f - 2.f * (qx * qx + qy * qy);

    float sx = scales[g * 3 + 0], sy = scales[g * 3 + 1], sz = scales[g * 3 + 2];
    float m00 = r00 * sx, m01 = r01 * sy, m02 = r02 * sz;
    float m10 = r10 * sx, m11 = r11 * sy, m12 = r12 * sz;
    float m20 = r20 * sx, m21 = r21 * sy, m22 = r22 * sz;

    float c00 = m00 * m00 + m01 * m01 + m02 * m02;
    float c01 = m00 * m10 + m01 * m11 + m02 * m12;
    float c02 = m00 * m20 + m01 * m21 + m02 * m22;
    float c11 = m10 * m10 + m11 * m11 + m12 * m12;
    float c12 = m10 * m20 + m11 * m21 + m12 * m22;
    float c22 = m20 * m20 + m21 * m21 + m22 * m22;

    float det = c00 * (c11 * c22 - c12 * c12)
              - c01 * (c01 * c22 - c12 * c02)
              + c02 * (c01 * c12 - c11 * c02);
    float id = 1.0f / det;
    float S00 = (c11 * c22 - c12 * c12) * id;
    float S01 = (c02 * c12 - c01 * c22) * id;
    float S02 = (c01 * c12 - c02 * c11) * id;
    float S11 = (c00 * c22 - c02 * c02) * id;
    float S12 = (c01 * c02 - c00 * c12) * id;
    float S22 = (c00 * c11 - c01 * c01) * id;

    float A0 = S00 * mx + S01 * my + S02 * mz;
    float A1 = S01 * mx + S11 * my + S12 * mz;
    float A2 = S02 * mx + S12 * my + S22 * mz;
    float cst = mx * A0 + my * A1 + mz * A2;

    float* o = gp + (size_t)g * 16;
    o[0] = S00; o[1] = S01; o[2] = S02; o[3]  = S11; o[4]  = S12; o[5]  = S22;
    o[6] = A0;  o[7] = A1;  o[8] = A2;  o[9]  = cst; o[10] = opac[g];
    o[11] = 0.f; o[12] = 0.f; o[13] = 0.f; o[14] = 0.f; o[15] = 0.f;
}

// ---------------------------------------------------------------------------
// Pre-swizzle features (f32 -> packed f16 pairs) into WMMA B-fragment layout.
// Layout: frag[(chunk*2 + ntile)*32 + lane][8 dwords].
// 16-bit B 32x16: lanes 0-15 carry N=0..15 with K pairs {0..7,16..23},
// lanes 16-31 carry N=0..15 with K pairs {8..15,24..31}.
// ---------------------------------------------------------------------------
__global__ __launch_bounds__(256) void prep_feat(
    const float* __restrict__ feat, unsigned int* __restrict__ bws)
{
    int f = blockIdx.x * 256 + threadIdx.x;          // 0..1023
    int c = f >> 6;                                  // chunk
    int t = (f >> 5) & 1;                            // n-tile
    int l = f & 31;                                  // lane
    int n = t * 16 + (l & 15);
    int klo = (l & 16) ? 8 : 0;
    unsigned int* dst = bws + (size_t)f * 8;
#pragma unroll
    for (int j = 0; j < 8; ++j) {
        int K = (j < 4) ? (klo + 2 * j) : (16 + klo + 2 * (j - 4));
        int g = c * 32 + K;
        dst[j] = pkh(feat[g * NFEAT + n], feat[(g + 1) * NFEAT + n]);
    }
}

// ---------------------------------------------------------------------------
// Main renderer: one wave = 32 points. Sequential transmittance chain in fp32
// VALU; weights@features via v_wmma_f32_16x16x32_f16 (4 per 32-Gaussian chunk).
// ---------------------------------------------------------------------------
__global__ __launch_bounds__(256) void gs_render(
    const float* __restrict__ coords,
    const float* __restrict__ gp,
    const unsigned int* __restrict__ bws,
    float* __restrict__ out)
{
    __shared__ __attribute__((aligned(16))) unsigned int ldsW[8 * 32 * ROWSTR];

    const int lane = threadIdx.x & 31;
    const int wave = threadIdx.x >> 5;
    const int pointBase = blockIdx.x * 256 + wave * 32;
    const int point = pointBase + lane;

    const float x = coords[(size_t)point * 3 + 0];
    const float y = coords[(size_t)point * 3 + 1];
    const float z = coords[(size_t)point * 3 + 2];

    unsigned int* wbase = ldsW + wave * (32 * ROWSTR);

    v8f c00 = {}; v8f c01 = {}; v8f c10 = {}; v8f c11 = {};
    float T = 1.0f;

    for (int c = 0; c < NCHUNK; ++c) {
        if (c + 1 < NCHUNK)
            __builtin_prefetch(bws + (size_t)((c + 1) * 2) * 32 * 8, 0, 1);

        // --- alpha / transmittance chain for this chunk (sequential in k) ---
        float w[32];
        const v4f* p = (const v4f*)gp + (size_t)(c * 32) * 4;
#pragma unroll
        for (int k = 0; k < 32; ++k) {
            v4f pa = p[k * 4 + 0];   // S00 S01 S02 S11
            v4f pb = p[k * 4 + 1];   // S12 S22 A0  A1
            v4f pc = p[k * 4 + 2];   // A2  cst op  pad
            float m = pa.x * x * x + pa.w * y * y + pb.y * z * z
                    + 2.0f * (pa.y * x * y + pa.z * x * z + pb.x * y * z)
                    - 2.0f * (x * pb.z + y * pb.w + z * pc.x) + pc.y;
            float alpha = pc.z * __expf(-0.5f * m);
            w[k] = T * alpha;
            T *= (1.0f - alpha);
        }

        // --- stage 32x32 weight tile in LDS (f16 pairs, row = my point) ---
        unsigned int* row = wbase + lane * ROWSTR;
#pragma unroll
        for (int d = 0; d < 4; ++d) {
            v4u q;
            q.x = pkh(w[8 * d + 0], w[8 * d + 1]);
            q.y = pkh(w[8 * d + 2], w[8 * d + 3]);
            q.z = pkh(w[8 * d + 4], w[8 * d + 5]);
            q.w = pkh(w[8 * d + 6], w[8 * d + 7]);
            *(v4u*)(row + 4 * d) = q;
        }

        // --- B fragments (precomputed layout) ---
        const v4u* b0p = (const v4u*)(bws + (size_t)((c * 2 + 0) * 32 + lane) * 8);
        const v4u* b1p = (const v4u*)(bws + (size_t)((c * 2 + 1) * 32 + lane) * 8);
        v4u b0l = b0p[0], b0h = b0p[1];
        v4u b1l = b1p[0], b1h = b1p[1];
        v8u bu0 = {b0l.x, b0l.y, b0l.z, b0l.w, b0h.x, b0h.y, b0h.z, b0h.w};
        v8u bu1 = {b1l.x, b1l.y, b1l.z, b1l.w, b1h.x, b1h.y, b1h.z, b1h.w};
        v16h B0 = __builtin_bit_cast(v16h, bu0);
        v16h B1 = __builtin_bit_cast(v16h, bu1);

        // --- A fragments from LDS: lanes 0-15 take K pairs {0..7,16..23},
        //     lanes 16-31 take K pairs {8..15,24..31} ---
        const int off = (lane & 16) ? 4 : 0;
        const int r0 = lane & 15;
        const unsigned int* a0p = wbase + r0 * ROWSTR + off;
        const unsigned int* a1p = wbase + (16 + r0) * ROWSTR + off;
        v4u a0l = *(const v4u*)(a0p);
        v4u a0h = *(const v4u*)(a0p + 8);
        v4u a1l = *(const v4u*)(a1p);
        v4u a1h = *(const v4u*)(a1p + 8);
        v8u au0 = {a0l.x, a0l.y, a0l.z, a0l.w, a0h.x, a0h.y, a0h.z, a0h.w};
        v8u au1 = {a1l.x, a1l.y, a1l.z, a1l.w, a1h.x, a1h.y, a1h.z, a1h.w};
        v16h A0 = __builtin_bit_cast(v16h, au0);
        v16h A1 = __builtin_bit_cast(v16h, au1);

        // --- D += A x B : 2 point-tiles x 2 feature-tiles, K = 32 ---
        c00 = __builtin_amdgcn_wmma_f32_16x16x32_f16(false, A0, false, B0, (short)0, c00, false, false);
        c01 = __builtin_amdgcn_wmma_f32_16x16x32_f16(false, A0, false, B1, (short)0, c01, false, false);
        c10 = __builtin_amdgcn_wmma_f32_16x16x32_f16(false, A1, false, B0, (short)0, c10, false, false);
        c11 = __builtin_amdgcn_wmma_f32_16x16x32_f16(false, A1, false, B1, (short)0, c11, false, false);
    }

    // --- store: 16x16 f32 D layout -> out[(point)*32 + feat] ---
    const int nn  = lane & 15;
    const int mhi = (lane >> 4) << 3;   // lanes 16-31 hold rows M+8
#pragma unroll
    for (int r = 0; r < 8; ++r) {
        int M = r + mhi;
        float* o0 = out + (size_t)(pointBase + M) * NFEAT;
        o0[nn]      = c00[r];
        o0[nn + 16] = c01[r];
        float* o1 = out + (size_t)(pointBase + 16 + M) * NFEAT;
        o1[nn]      = c10[r];
        o1[nn + 16] = c11[r];
    }
}

extern "C" void kernel_launch(void* const* d_in, const int* in_sizes, int n_in,
                              void* d_out, int out_size, void* d_ws, size_t ws_size,
                              hipStream_t stream) {
    const float* means  = (const float*)d_in[0];
    const float* scales = (const float*)d_in[1];
    const float* rots   = (const float*)d_in[2];
    const float* opac   = (const float*)d_in[3];
    const float* feat   = (const float*)d_in[4];
    const float* cam    = (const float*)d_in[5];
    const float* coords = (const float*)d_in[6];
    float* out = (float*)d_out;

    float* gp = (float*)d_ws;                                        // 32 KB
    unsigned int* bws = (unsigned int*)((char*)d_ws + NGAUSS * 16 * sizeof(float)); // 32 KB

    prep_gauss<<<2, 256, 0, stream>>>(means, scales, rots, opac, cam, gp);
    prep_feat<<<4, 256, 0, stream>>>(feat, bws);
    gs_render<<<NPOINTS / 256, 256, 0, stream>>>(coords, gp, bws, out);
}
